// Model_15083925144048
// MI455X (gfx1250) — matrix operand order repaired and checked by reading, not compiled, not run on hardware
//
#include <hip/hip_runtime.h>

typedef __attribute__((ext_vector_type(16))) _Float16 v16h;
typedef __attribute__((ext_vector_type(8)))  float    v8f;

#define B_  2048
#define T_  512
#define A_  32
#define H_  16
#define LDSW 24   // halves per LDS row: 48B stride -> 16B aligned reads, conflict-free banks

__device__ __forceinline__ float sigmoid_f(float x) {
    return 1.0f / (1.0f + __expf(-x));
}
__device__ __forceinline__ float tanh_f(float x) {
    // tanh(x) = 1 - 2/(exp(2x)+1); correct limits at +/-inf
    return 1.0f - 2.0f / (__expf(2.0f * x) + 1.0f);
}

__device__ __forceinline__ void pack8(v16h& d, int base, float4 a, float4 b) {
    d[base + 0] = (_Float16)a.x; d[base + 1] = (_Float16)a.y;
    d[base + 2] = (_Float16)a.z; d[base + 3] = (_Float16)a.w;
    d[base + 4] = (_Float16)b.x; d[base + 5] = (_Float16)b.y;
    d[base + 6] = (_Float16)b.z; d[base + 7] = (_Float16)b.w;
}

__global__ __launch_bounds__(32)
void lstm_ac_scan_kernel(const float* __restrict__ inputs,   // [B,T,A]
                         const float* __restrict__ masks,    // [B,T]
                         const float* __restrict__ h0,       // [1,B,H]
                         const float* __restrict__ c0,       // [1,B,H]
                         const float* __restrict__ w_ih,     // [4H,A]
                         const float* __restrict__ w_hh,     // [4H,H]
                         const float* __restrict__ b_ih,     // [4H]
                         const float* __restrict__ b_hh,     // [4H]
                         const float* __restrict__ w_actor,  // [A,H]
                         const float* __restrict__ b_actor,  // [A]
                         const float* __restrict__ w_critic, // [1,H]
                         const float* __restrict__ b_critic, // [1]
                         float* __restrict__ actor_out,      // [B,T,A]
                         float* __restrict__ critic_out,     // [B,T]
                         float* __restrict__ hT_out,         // [B,H]
                         float* __restrict__ cT_out)         // [B,H]
{
    __shared__ alignas(16) _Float16 hbuf[16 * LDSW];

    const int  lane = threadIdx.x & 31;
    const int  rowl = lane & 15;          // A-layout row / D-layout column
    const bool hi   = lane >= 16;
    const int  off8 = hi ? 8 : 0;         // A-layout K base (16-bit), D-layout row base
    const int  kb16 = hi ? 16 : 0;        // unused; B fragments use off8 as K base
    const int  b0   = blockIdx.x * 16;    // batch tile base

    // ---- resident weight fragments (B-matrix layout: lane holds column n, K=off8..off8+7 and 16+off8..16+off8+7) ----
    v16h wih[4], whh[4], wa[2];
    float bsum[4], ba[2];
#pragma unroll
    for (int g = 0; g < 4; ++g) {
        const int n = g * 16 + rowl;
        const float* wr = w_ih + (size_t)n * A_ + off8;
#pragma unroll
        for (int e = 0; e < 16; ++e) wih[g][e] = (_Float16)wr[e + (e >= 8 ? 8 : 0)];
        const float* hr = w_hh + (size_t)n * H_;
#pragma unroll
        for (int e = 0; e < 16; ++e) whh[g][e] = (e >= 8) ? (_Float16)0.0f : (_Float16)hr[off8 + e];
        bsum[g] = b_ih[n] + b_hh[n];
    }
#pragma unroll
    for (int t2 = 0; t2 < 2; ++t2) {
        const int n = t2 * 16 + rowl;
        const float* ar = w_actor + (size_t)n * H_;
#pragma unroll
        for (int e = 0; e < 16; ++e) wa[t2][e] = (e >= 8) ? (_Float16)0.0f : (_Float16)ar[off8 + e];
        ba[t2] = b_actor[n];
    }
    float wcl[8];
#pragma unroll
    for (int j = 0; j < 8; ++j) wcl[j] = w_critic[off8 + j];
    const float bc = b_critic[0];

    // ---- state init (D layout: element r -> (row r+off8, col rowl)) ----
    v8f hst, cst;
#pragma unroll
    for (int r = 0; r < 8; ++r) {
        const int m = r + off8;
        hst[r] = h0[(size_t)(b0 + m) * H_ + rowl];
        cst[r] = c0[(size_t)(b0 + m) * H_ + rowl];
        hbuf[m * LDSW + rowl] = (_Float16)hst[r];
    }
    __syncthreads();

    // A-layout h fragment: lane holds row rowl, K = off8..off8+7 (K>=16 zero-padded)
    v16h ahraw;
    {
        union { float4 f; _Float16 h[8]; } u;
        u.f = *(const float4*)(&hbuf[rowl * LDSW + off8]);
#pragma unroll
        for (int j = 0; j < 8; ++j) ahraw[j] = u.h[j];
#pragma unroll
        for (int j = 8; j < 16; ++j) ahraw[j] = (_Float16)0.0f;
    }
    __syncthreads();

    const float* xbase = inputs + (size_t)(b0 + rowl) * T_ * A_;
    const float* mbase = masks  + (size_t)(b0 + rowl) * T_;
    const v8f vzero = {0.f, 0.f, 0.f, 0.f, 0.f, 0.f, 0.f, 0.f};

    for (int t = 0; t < T_; ++t) {
        // episode-reset mask: h,c scaled before the step
        const float mA = mbase[t];
        v16h ah = ahraw;
        const _Float16 mh = (_Float16)mA;
#pragma unroll
        for (int j = 0; j < 8; ++j) ah[j] = ah[j] * mh;
#pragma unroll
        for (int r = 0; r < 8; ++r) {
            const float mr = __shfl(mA, r + off8, 32);
            cst[r] *= mr;
        }

        // x fragment (A layout, K=0..31)
        const float* xr = xbase + (size_t)t * A_;
        __builtin_prefetch(xr + A_, 0, 3);
        float4 x0 = *(const float4*)(xr + off8);
        float4 x1 = *(const float4*)(xr + off8 + 4);
        float4 x2 = *(const float4*)(xr + 16 + off8);
        float4 x3 = *(const float4*)(xr + 16 + off8 + 4);
        v16h ax;
        pack8(ax, 0, x0, x1);
        pack8(ax, 8, x2, x3);

        // gates = x@Wih^T + h@Whh^T (bias added in activation pass)
        v8f gi = __builtin_amdgcn_wmma_f32_16x16x32_f16(false, ax, false, wih[0], (short)0, vzero, false, false);
        gi     = __builtin_amdgcn_wmma_f32_16x16x32_f16(false, ah, false, whh[0], (short)0, gi,    false, false);
        v8f gf = __builtin_amdgcn_wmma_f32_16x16x32_f16(false, ax, false, wih[1], (short)0, vzero, false, false);
        gf     = __builtin_amdgcn_wmma_f32_16x16x32_f16(false, ah, false, whh[1], (short)0, gf,    false, false);
        v8f gg = __builtin_amdgcn_wmma_f32_16x16x32_f16(false, ax, false, wih[2], (short)0, vzero, false, false);
        gg     = __builtin_amdgcn_wmma_f32_16x16x32_f16(false, ah, false, whh[2], (short)0, gg,    false, false);
        v8f go = __builtin_amdgcn_wmma_f32_16x16x32_f16(false, ax, false, wih[3], (short)0, vzero, false, false);
        go     = __builtin_amdgcn_wmma_f32_16x16x32_f16(false, ah, false, whh[3], (short)0, go,    false, false);

        // activations + state update (D layout), recirculate h through LDS
#pragma unroll
        for (int r = 0; r < 8; ++r) {
            const float iv = sigmoid_f(gi[r] + bsum[0]);
            const float fv = sigmoid_f(gf[r] + bsum[1]);
            const float gv = tanh_f   (gg[r] + bsum[2]);
            const float ov = sigmoid_f(go[r] + bsum[3]);
            const float cn = fv * cst[r] + iv * gv;
            const float hn = ov * tanh_f(cn);
            cst[r] = cn;
            hst[r] = hn;
            hbuf[(r + off8) * LDSW + rowl] = (_Float16)hn;
        }
        __syncthreads();
        {
            union { float4 f; _Float16 h[8]; } u;
            u.f = *(const float4*)(&hbuf[rowl * LDSW + off8]);
#pragma unroll
            for (int j = 0; j < 8; ++j) ahraw[j] = u.h[j];
        }
        __syncthreads();

        // fused actor head: feats@Wa^T (+ bias at store)
#pragma unroll
        for (int t2 = 0; t2 < 2; ++t2) {
            v8f aacc = __builtin_amdgcn_wmma_f32_16x16x32_f16(false, ahraw, false, wa[t2], (short)0, vzero, false, false);
#pragma unroll
            for (int r = 0; r < 8; ++r) {
                actor_out[((size_t)(b0 + r + off8) * T_ + t) * A_ + t2 * 16 + rowl] = aacc[r] + ba[t2];
            }
        }

        // fused critic head: per-lane partial dot over K=off8..off8+7, pair-reduce across halves
        float part = 0.0f;
#pragma unroll
        for (int j = 0; j < 8; ++j) part += (float)ahraw[j] * wcl[j];
        part += __shfl_xor(part, 16, 32);
        if (!hi) critic_out[(size_t)(b0 + rowl) * T_ + t] = part + bc;
    }

    // final states
#pragma unroll
    for (int r = 0; r < 8; ++r) {
        const int m = r + off8;
        hT_out[(size_t)(b0 + m) * H_ + rowl] = hst[r];
        cT_out[(size_t)(b0 + m) * H_ + rowl] = cst[r];
    }
}

extern "C" void kernel_launch(void* const* d_in, const int* in_sizes, int n_in,
                              void* d_out, int out_size, void* d_ws, size_t ws_size,
                              hipStream_t stream) {
    (void)in_sizes; (void)n_in; (void)out_size; (void)d_ws; (void)ws_size;

    const float* inputs   = (const float*)d_in[0];
    const float* masks    = (const float*)d_in[1];
    const float* h0       = (const float*)d_in[2];
    const float* c0       = (const float*)d_in[3];
    const float* w_ih     = (const float*)d_in[4];
    const float* w_hh     = (const float*)d_in[5];
    const float* b_ih     = (const float*)d_in[6];
    const float* b_hh     = (const float*)d_in[7];
    const float* w_actor  = (const float*)d_in[8];
    const float* b_actor  = (const float*)d_in[9];
    const float* w_critic = (const float*)d_in[10];
    const float* b_critic = (const float*)d_in[11];

    float* out       = (float*)d_out;
    float* actor     = out;                                   // B*T*A
    float* critic    = actor  + (size_t)B_ * T_ * A_;         // B*T
    float* hT        = critic + (size_t)B_ * T_;              // B*H
    float* cT        = hT     + (size_t)B_ * H_;              // B*H

    dim3 grid(B_ / 16);
    dim3 block(32);
    lstm_ac_scan_kernel<<<grid, block, 0, stream>>>(
        inputs, masks, h0, c0, w_ih, w_hh, b_ih, b_hh,
        w_actor, b_actor, w_critic, b_critic,
        actor, critic, hT, cT);
}
